// SSDV_44478681317688
// MI455X (gfx1250) — compile-verified, matrix operand order checked by reading, not run on hardware
//
#include <hip/hip_runtime.h>

// ---------------------------------------------------------------------------
// Mamba-2 / SSD forward for gfx1250 (MI455X).
//   - bf16 WMMA (v_wmma_f32_16x16x32_bf16) for the two big projections,
//     with the B-panel staged in LDS by the Tensor Data Mover
//     (tensor_load_to_lds + s_wait_tensorcnt, double buffered).
//   - fp32 WMMA (v_wmma_f32_16x16x4_f32) for the SSD-core einsums.
// ---------------------------------------------------------------------------

typedef unsigned short u16;
typedef unsigned int   u32;

typedef __attribute__((ext_vector_type(16))) __bf16 v16bf;
typedef __attribute__((ext_vector_type(8)))  float  v8f;
typedef __attribute__((ext_vector_type(2)))  float  v2f;
typedef __attribute__((ext_vector_type(4)))  u32    v4u;
typedef __attribute__((ext_vector_type(8)))  u32    v8u;

#define NFEAT   1024
#define NHEADS  8
#define NSTATES 64
#define HALFD   1024
#define HEADDIM 128
#define CONVD   1152
#define NPROJ   2184      // HALF + CONVDIM + HEADS
#define NPROJP  2192      // padded to multiple of 16
#define SEQL    4096
#define NB      2
#define NTOK    (NB*SEQL) // 8192
#define CHUNKL  256
#define NCHUNK  16
#define INNERD  2048

#define KCH     512             // K-chunk of the B panel staged in LDS
#define BROWP   (KCH + 16)      // LDS row pitch in u16 (=1056B: +8 dwords pad
                                // per row -> 8-bank shift, conflict-free frags)

// ---- fp32 -> bf16 (round to nearest even) ---------------------------------
__device__ __forceinline__ u16 f2bf(float f) {
  u32 u = __builtin_bit_cast(u32, f);
  u32 r = u + 0x7FFFu + ((u >> 16) & 1u);
  return (u16)(r >> 16);
}

struct U8x4 { uint4 lo, hi; };

// Load a 16-element bf16 fragment (A or B operand of 16x16x32):
// lane(g,m): elements 0..7  = K = k0 + 8g .. +7
//            elements 8..15 = K = k0 + 16 + 8g .. +7
__device__ __forceinline__ v16bf load_bf16_frag(const u16* base, int row, int ld,
                                                int k0, int g) {
  const u16* p = base + (size_t)row * ld + k0 + 8 * g;
  U8x4 u;
  u.lo = *(const uint4*)(p);
  u.hi = *(const uint4*)(p + 16);
  return __builtin_bit_cast(v16bf, u);
}

// ---------------------------------------------------------------------------
// Tensor Data Mover: DMA a 16-row x kch-col bf16 tile (row stride K elements)
// from global into LDS at lds_off, with 8-DWORD padding after each 1KB row.
// D# layout per CDNA5 ISA.  Uniform operands ->
// SGPR groups; issued by one wave, completion via TENSORcnt.
// ---------------------------------------------------------------------------
__device__ __forceinline__ void tdm_load_tile16(u32 lds_off, const u16* gsrc,
                                                int K, int kch) {
  unsigned long long ga = (unsigned long long)(__UINTPTR_TYPE__)gsrc;
  v4u g0;
  g0[0] = 1u;                                               // count=1, user mode
  g0[1] = lds_off;                                          // lds_addr (bytes)
  g0[2] = (u32)ga;                                          // global_addr lo32
  g0[3] = (u32)((ga >> 32) & 0x01FFFFFFull) | 0x80000000u;  // addr hi25 | type=2
  v8u g1;
  g1[0] = (1u << 16)        // data_size = 1 (2 bytes)
        | (1u << 20)        // pad_enable
        | (7u << 22)        // pad_interval = 256 DWORDs (one 1KB row)
        | (7u << 25);       // pad_amount   = 8 DWORDs (32B)
  g1[1] = ((u32)K & 0xFFFFu) << 16;              // tensor_dim0[15:0]
  g1[2] = ((u32)K >> 16) | (16u << 16);          // tensor_dim0[31:16] | tensor_dim1 lo
  g1[3] = ((u32)kch << 16);                      // tensor_dim1 hi (0) | tile_dim0
  g1[4] = 16u;                                   // tile_dim1 = 16, tile_dim2 = 0
  g1[5] = (u32)K;                                // tensor_dim0_stride lo32
  g1[6] = 0u;
  g1[7] = 0u;
  v4u gz = {0u, 0u, 0u, 0u};                     // groups 2/3: dims >= 2 unused
  asm volatile("tensor_load_to_lds %0, %1, %2, %3"
               :: "s"(g0), "s"(g1), "s"(gz), "s"(gz)
               : "memory");
}

// ---------------------------------------------------------------------------
// Conversion kernels
// ---------------------------------------------------------------------------
__global__ void cvt_bf16_kernel(const float* __restrict__ in, u16* __restrict__ out, int n) {
  int i = blockIdx.x * 256 + threadIdx.x;
  if (i < n) out[i] = f2bf(in[i]);
}

// out[n][k] = in[k][n]   (out: [cols_pad][rows], zero fill for n >= cols_in)
__global__ void transpose_cvt_kernel(const float* __restrict__ in, u16* __restrict__ out,
                                     int rows, int cols_in, int cols_pad) {
  int idx = blockIdx.x * 256 + threadIdx.x;
  if (idx >= cols_pad * rows) return;
  int k = idx % rows;
  int n = idx / rows;
  float v = (n < cols_in) ? in[(size_t)k * cols_in + n] : 0.0f;
  out[idx] = f2bf(v);
}

// ---------------------------------------------------------------------------
// bf16 GEMM:  C[M][Nreal] (fp32) = A[M][K] (bf16) * BT[Npad][K]^T (bf16)
// wave computes 64x16, block (8 waves) computes 512x16.
// B panel (16 x K) staged in LDS by the TDM, double buffered; DMA of chunk
// i+1 overlaps the WMMAs of chunk i.  grid.x = Npad/16, grid.y = M/512.
// ---------------------------------------------------------------------------
__global__ void gemm_bf16_nt(const u16* __restrict__ A, const u16* __restrict__ BT,
                             float* __restrict__ C, int M, int Nreal, int K, int ldc) {
  __shared__ u16 Bs[2][16 * BROWP];

  int lane = threadIdx.x & 31;
  int wave = threadIdx.x >> 5;
  int g = lane >> 4, m = lane & 15;
  int nbase = blockIdx.x * 16;
  int mbase = (blockIdx.y * 8 + wave) * 64;

  v8f acc0 = {0,0,0,0,0,0,0,0};
  v8f acc1 = acc0, acc2 = acc0, acc3 = acc0;

  const u16* bpanel = BT + (size_t)nbase * K;
  int nchunks = K / KCH;

  if (wave == 0)
    tdm_load_tile16((u32)(__UINTPTR_TYPE__)&Bs[0][0], bpanel, K, KCH);

  for (int ci = 0; ci < nchunks; ++ci) {
    if (wave == 0) __builtin_amdgcn_s_wait_tensorcnt(0);
    __syncthreads();                      // chunk ci resident for all waves
    if (wave == 0 && ci + 1 < nchunks)    // prefetch next chunk into other buffer
      tdm_load_tile16((u32)(__UINTPTR_TYPE__)&Bs[(ci + 1) & 1][0],
                      bpanel + (size_t)(ci + 1) * KCH, K, KCH);

    const u16* bs = &Bs[ci & 1][0];
    int kg = ci * KCH;
    for (int k0 = 0; k0 < KCH; k0 += 32) {
      v16bf bfr = load_bf16_frag(bs, m, BROWP, k0, g);           // ds_load_b128
      v16bf a0 = load_bf16_frag(A, mbase +  0 + m, K, kg + k0, g);
      v16bf a1 = load_bf16_frag(A, mbase + 16 + m, K, kg + k0, g);
      v16bf a2 = load_bf16_frag(A, mbase + 32 + m, K, kg + k0, g);
      v16bf a3 = load_bf16_frag(A, mbase + 48 + m, K, kg + k0, g);
      acc0 = __builtin_amdgcn_wmma_f32_16x16x32_bf16(false, a0, false, bfr, (short)0, acc0, false, false);
      acc1 = __builtin_amdgcn_wmma_f32_16x16x32_bf16(false, a1, false, bfr, (short)0, acc1, false, false);
      acc2 = __builtin_amdgcn_wmma_f32_16x16x32_bf16(false, a2, false, bfr, (short)0, acc2, false, false);
      acc3 = __builtin_amdgcn_wmma_f32_16x16x32_bf16(false, a3, false, bfr, (short)0, acc3, false, false);
    }
  }

  int col = nbase + m;
  if (col < Nreal) {
    v8f accs[4] = {acc0, acc1, acc2, acc3};
    #pragma unroll
    for (int i = 0; i < 4; ++i)
      #pragma unroll
      for (int r = 0; r < 8; ++r) {
        int row = mbase + 16 * i + r + 8 * g;
        C[(size_t)row * ldc + col] = accs[i][r];
      }
  }
  (void)M;
}

// ---------------------------------------------------------------------------
// dt = softplus(zx[:, 2176:2184] + dt_bias)
// ---------------------------------------------------------------------------
__global__ void dt_softplus_kernel(const float* __restrict__ zx,
                                   const float* __restrict__ dt_bias,
                                   float* __restrict__ dt) {
  int i = blockIdx.x * 256 + threadIdx.x;
  if (i >= NTOK * NHEADS) return;
  int t = i >> 3, h = i & 7;
  float v = zx[(size_t)t * NPROJ + (HALFD + CONVD) + h] + dt_bias[h];
  dt[i] = (v > 20.0f) ? v : log1pf(__expf(v));
}

// ---------------------------------------------------------------------------
// causal depthwise conv (width 4) + SiLU, reading channels [in_off, in_off+C)
// of zx, writing out[t][c].
// ---------------------------------------------------------------------------
__global__ void dwconv_silu_kernel(const float* __restrict__ zx,
                                   const float* __restrict__ w,
                                   const float* __restrict__ bias,
                                   float* __restrict__ out, int in_off, int C) {
  int idx = blockIdx.x * 256 + threadIdx.x;
  if (idx >= NTOK * C) return;
  int c = idx % C;
  int t = idx / C;
  int tl = t & (SEQL - 1);
  float acc = bias[c];
  #pragma unroll
  for (int j = 0; j < 4; ++j) {
    int tt = tl - 3 + j;
    if (tt >= 0)
      acc += zx[(size_t)(t - tl + tt) * NPROJ + in_off + c] * w[c * 4 + j];
  }
  out[(size_t)t * C + c] = acc / (1.0f + __expf(-acc));
}

// ---------------------------------------------------------------------------
// SSD intra-chunk kernel: one block per (b, chunk, head).
//   - cumsum of A*dt (Hillis-Steele in LDS), written to cs_g
//   - Y_diag = (tril(exp(cs_l - cs_s)) .* (C B^T) .* dt_s) @ x   (+ D*x)
//   - states = (x .* dt .* exp(cs_end - cs))^T @ B
// fp32 WMMA 16x16x4 throughout.
// ---------------------------------------------------------------------------
__device__ __forceinline__ float xcv(const float* xc, size_t tbase, int s, int cidx) {
  return xc[(tbase + (size_t)s) * CONVD + cidx];
}

__global__ void ssd_chunk_kernel(const float* __restrict__ xc,
                                 const float* __restrict__ dtg,
                                 const float* __restrict__ Alog,
                                 const float* __restrict__ Dp,
                                 float* __restrict__ cs_g,
                                 float* __restrict__ states_g,
                                 float* __restrict__ yf) {
  int blk = blockIdx.x;          // ((b*16 + c)*8 + h)
  int h = blk & 7;
  int c = (blk >> 3) & 15;
  int b = blk >> 7;
  int tid = threadIdx.x;
  int lane = tid & 31, wave = tid >> 5;
  int g = lane >> 4, m = lane & 15;

  __shared__ float dts[256], csA[256], csB[256], ws[256];
  __shared__ float Gbuf[8][16][16];

  size_t tbase = ((size_t)b * NCHUNK + c) * CHUNKL;
  float Aneg = -__expf(Alog[h]);
  float dtv = dtg[(tbase + tid) * NHEADS + h];
  dts[tid] = dtv;
  csA[tid] = dtv * Aneg;
  __syncthreads();

  // inclusive scan over 256 elements
  float* sc = csA;
  float* dc = csB;
  for (int off = 1; off < 256; off <<= 1) {
    float v = sc[tid];
    if (tid >= off) v += sc[tid - off];
    __syncthreads();
    dc[tid] = v;
    __syncthreads();
    float* tp = sc; sc = dc; dc = tp;
  }
  float cst = sc[255];
  float csv = sc[tid];
  __syncthreads();
  csA[tid] = csv;
  ws[tid] = dts[tid] * __expf(cst - csv);
  cs_g[((size_t)(b * NHEADS + h) * NCHUNK + c) * CHUNKL + tid] = csv;
  __syncthreads();

  float Dh = Dp[h];

  // ---- Phase 1: Y_diag, two l-tiles per wave (balanced: w and 15-w) ----
  for (int pass = 0; pass < 2; ++pass) {
    int lt = (pass == 0) ? wave : (15 - wave);
    int lbase = lt * 16;
    v8f Y[8];
    #pragma unroll
    for (int i = 0; i < 8; ++i) Y[i] = (v8f){0,0,0,0,0,0,0,0};

    for (int st = 0; st <= lt; ++st) {
      int sbase = st * 16;
      // S = C(l,:) @ B(s,:)^T over 64 states
      v8f S = {0,0,0,0,0,0,0,0};
      #pragma unroll
      for (int k0 = 0; k0 < 64; k0 += 4) {
        int kk = k0 + 2 * g;
        v2f a, bb;
        a[0]  = xcv(xc, tbase, lbase + m, HALFD + NSTATES + kk);
        a[1]  = xcv(xc, tbase, lbase + m, HALFD + NSTATES + kk + 1);
        bb[0] = xcv(xc, tbase, sbase + m, HALFD + kk);
        bb[1] = xcv(xc, tbase, sbase + m, HALFD + kk + 1);
        S = __builtin_amdgcn_wmma_f32_16x16x4_f32(false, a, false, bb, (short)0, S, false, false);
      }
      // mask / decay / dt_s, stage G into per-wave LDS tile
      int sg = sbase + m;
      float cs_s = csA[sg];
      float dt_s = dts[sg];
      #pragma unroll
      for (int r = 0; r < 8; ++r) {
        int lg = lbase + r + 8 * g;
        float v = (lg >= sg) ? S[r] * __expf(csA[lg] - cs_s) * dt_s : 0.0f;
        Gbuf[wave][r + 8 * g][m] = v;
      }
      // Y += G(16x16) @ x(16x128)   (per-wave LDS RAW is in-order)
      #pragma unroll
      for (int k0 = 0; k0 < 16; k0 += 4) {
        int kk = k0 + 2 * g;
        v2f a;
        a[0] = Gbuf[wave][m][kk];
        a[1] = Gbuf[wave][m][kk + 1];
        #pragma unroll
        for (int pt = 0; pt < 8; ++pt) {
          v2f bb;
          bb[0] = xcv(xc, tbase, sbase + kk,     h * HEADDIM + pt * 16 + m);
          bb[1] = xcv(xc, tbase, sbase + kk + 1, h * HEADDIM + pt * 16 + m);
          Y[pt] = __builtin_amdgcn_wmma_f32_16x16x4_f32(false, a, false, bb, (short)0, Y[pt], false, false);
        }
      }
    }
    // + D*x, store Y
    #pragma unroll
    for (int pt = 0; pt < 8; ++pt) {
      int p = pt * 16 + m;
      #pragma unroll
      for (int r = 0; r < 8; ++r) {
        int lg = lbase + r + 8 * g;
        float out = Y[pt][r] + Dh * xcv(xc, tbase, lg, h * HEADDIM + p);
        yf[(tbase + lg) * (size_t)HALFD + h * HEADDIM + p] = out;
      }
    }
  }

  // ---- Phase 2: chunk states[p][n] = sum_s x[s,p]*ws[s]*B[s,n] ----
  {
    int pbase = wave * 16;
    float* sp = states_g + (((size_t)(b * NCHUNK + c) * NHEADS + h) * HEADDIM) * NSTATES;
    for (int nt = 0; nt < 4; ++nt) {
      int nbase = nt * 16;
      v8f acc = {0,0,0,0,0,0,0,0};
      for (int k0 = 0; k0 < 256; k0 += 4) {
        int kk = k0 + 2 * g;
        v2f a, bb;
        a[0]  = xcv(xc, tbase, kk,     h * HEADDIM + pbase + m) * ws[kk];
        a[1]  = xcv(xc, tbase, kk + 1, h * HEADDIM + pbase + m) * ws[kk + 1];
        bb[0] = xcv(xc, tbase, kk,     HALFD + nbase + m);
        bb[1] = xcv(xc, tbase, kk + 1, HALFD + nbase + m);
        acc = __builtin_amdgcn_wmma_f32_16x16x4_f32(false, a, false, bb, (short)0, acc, false, false);
      }
      #pragma unroll
      for (int r = 0; r < 8; ++r) {
        int p = pbase + r + 8 * g;
        sp[(size_t)p * NSTATES + nbase + m] = acc[r];
      }
    }
  }
}

// ---------------------------------------------------------------------------
// Sequential inter-chunk scan (16 chunks): prev_c = carry; carry = states_c + cd_c*carry
// ---------------------------------------------------------------------------
__global__ void ssd_scan_kernel(const float* __restrict__ states,
                                const float* __restrict__ cs_g,
                                float* __restrict__ prev) {
  int idx = blockIdx.x * 256 + threadIdx.x;
  if (idx >= NB * NHEADS * HEADDIM * NSTATES) return;
  int n = idx & 63;
  int p = (idx >> 6) & 127;
  int h = (idx >> 13) & 7;
  int b = idx >> 16;
  float carry = 0.0f;
  for (int c = 0; c < NCHUNK; ++c) {
    size_t off = (((size_t)(b * NCHUNK + c) * NHEADS + h) * HEADDIM + p) * NSTATES + n;
    prev[off] = carry;
    float cd = __expf(cs_g[((size_t)(b * NHEADS + h) * NCHUNK + c) * CHUNKL + (CHUNKL - 1)]);
    carry = states[off] + cd * carry;
  }
}

// ---------------------------------------------------------------------------
// Y_off = exp(cs_l) * C(l,:) @ prev(p,:)^T, accumulated into yf. Block per (b,c,h).
// ---------------------------------------------------------------------------
__global__ void ssd_yoff_kernel(const float* __restrict__ xc,
                                const float* __restrict__ cs_g,
                                const float* __restrict__ prev,
                                float* __restrict__ yf) {
  int blk = blockIdx.x;
  int h = blk & 7;
  int c = (blk >> 3) & 15;
  int b = blk >> 7;
  int tid = threadIdx.x;
  int lane = tid & 31, wave = tid >> 5;
  int g = lane >> 4, m = lane & 15;

  __shared__ float cse[256];
  size_t tbase = ((size_t)b * NCHUNK + c) * CHUNKL;
  cse[tid] = __expf(cs_g[((size_t)(b * NHEADS + h) * NCHUNK + c) * CHUNKL + tid]);
  __syncthreads();

  const float* pv = prev + ((size_t)(b * NCHUNK + c) * NHEADS + h) * HEADDIM * NSTATES;

  for (int pass = 0; pass < 2; ++pass) {
    int lt = wave + (pass ? 8 : 0);
    int lbase = lt * 16;
    v8f Y[8];
    #pragma unroll
    for (int i = 0; i < 8; ++i) Y[i] = (v8f){0,0,0,0,0,0,0,0};

    #pragma unroll
    for (int k0 = 0; k0 < 64; k0 += 4) {
      int kk = k0 + 2 * g;
      v2f a;
      a[0] = xcv(xc, tbase, lbase + m, HALFD + NSTATES + kk);
      a[1] = xcv(xc, tbase, lbase + m, HALFD + NSTATES + kk + 1);
      #pragma unroll
      for (int pt = 0; pt < 8; ++pt) {
        v2f bb;
        bb[0] = pv[(size_t)(pt * 16 + m) * NSTATES + kk];
        bb[1] = pv[(size_t)(pt * 16 + m) * NSTATES + kk + 1];
        Y[pt] = __builtin_amdgcn_wmma_f32_16x16x4_f32(false, a, false, bb, (short)0, Y[pt], false, false);
      }
    }
    #pragma unroll
    for (int pt = 0; pt < 8; ++pt) {
      int p = pt * 16 + m;
      #pragma unroll
      for (int r = 0; r < 8; ++r) {
        int lg = lbase + r + 8 * g;
        size_t o = (tbase + lg) * (size_t)HALFD + h * HEADDIM + p;
        yf[o] += Y[pt][r] * cse[lg];
      }
    }
  }
}

// ---------------------------------------------------------------------------
// RMS-norm over concat([y, z], dim=-1) (2048), scale by norm_w, emit bf16.
// One block per token.
// ---------------------------------------------------------------------------
__global__ void rmsnorm_cat_kernel(const float* __restrict__ yf,
                                   const float* __restrict__ zc,
                                   const float* __restrict__ norm_w,
                                   u16* __restrict__ y16) {
  int t = blockIdx.x;
  int tid = threadIdx.x;
  __shared__ float red[256];
  float vals[8];
  float s = 0.0f;
  #pragma unroll
  for (int k = 0; k < 8; ++k) {
    int j = tid + k * 256;
    float v = (j < HALFD) ? yf[(size_t)t * HALFD + j]
                          : zc[(size_t)t * HALFD + (j - HALFD)];
    vals[k] = v;
    s += v * v;
  }
  red[tid] = s;
  __syncthreads();
  for (int off = 128; off > 0; off >>= 1) {
    if (tid < off) red[tid] += red[tid + off];
    __syncthreads();
  }
  float rinv = rsqrtf(red[0] / (float)INNERD + 1e-5f);
  #pragma unroll
  for (int k = 0; k < 8; ++k) {
    int j = tid + k * 256;
    y16[(size_t)t * INNERD + j] = f2bf(vals[k] * rinv * norm_w[j]);
  }
}

// ---------------------------------------------------------------------------
// Host-side orchestration
// ---------------------------------------------------------------------------
extern "C" void kernel_launch(void* const* d_in, const int* in_sizes, int n_in,
                              void* d_out, int out_size, void* d_ws, size_t ws_size,
                              hipStream_t stream) {
  const float* u       = (const float*)d_in[0];
  const float* W_in    = (const float*)d_in[1];
  const float* xconv_w = (const float*)d_in[2];
  const float* xconv_b = (const float*)d_in[3];
  const float* zconv_w = (const float*)d_in[4];
  const float* zconv_b = (const float*)d_in[5];
  const float* dt_bias = (const float*)d_in[6];
  const float* Alog    = (const float*)d_in[7];
  const float* Dp      = (const float*)d_in[8];
  const float* norm_w  = (const float*)d_in[9];
  const float* W_out   = (const float*)d_in[10];
  float* out = (float*)d_out;

  char* ws = (char*)d_ws;
  size_t o = 0;
  u16*   u16b = (u16*)(ws + o);   o += (size_t)NTOK * NFEAT * 2;        // 16 MB
  u16*   wt   = (u16*)(ws + o);   o += (size_t)NPROJP * NFEAT * 2;      // 4.5 MB
  float* zx   = (float*)(ws + o); o += (size_t)NTOK * NPROJ * 4;        // 71.6 MB
  float* xcb  = (float*)(ws + o); o += (size_t)NTOK * CONVD * 4;        // 37.7 MB
  float* zcb  = (float*)(ws + o); o += (size_t)NTOK * HALFD * 4;        // 33.6 MB
  float* dtb  = (float*)(ws + o); o += (size_t)NTOK * NHEADS * 4;       // 0.25 MB
  float* csb  = (float*)(ws + o); o += (size_t)NB * NHEADS * NCHUNK * CHUNKL * 4; // 0.25 MB
  float* stb  = (float*)(ws + o); o += (size_t)NB * NCHUNK * NHEADS * HEADDIM * NSTATES * 4; // 8.4 MB
  float* pvb  = (float*)(ws + o); o += (size_t)NB * NCHUNK * NHEADS * HEADDIM * NSTATES * 4; // 8.4 MB
  float* yfb  = (float*)(ws + o); o += (size_t)NTOK * HALFD * 4;        // 33.6 MB
  u16*   y16b = (u16*)(ws + o);   o += (size_t)NTOK * INNERD * 2;       // 33.6 MB
  u16*   wot  = (u16*)(ws + o);   o += (size_t)NFEAT * INNERD * 2;      // 4.2 MB

  // 1) convert activations / weights to bf16 (W transposed to [N][K])
  cvt_bf16_kernel<<<(NTOK * NFEAT + 255) / 256, 256, 0, stream>>>(u, u16b, NTOK * NFEAT);
  transpose_cvt_kernel<<<(NPROJP * NFEAT + 255) / 256, 256, 0, stream>>>(W_in, wt, NFEAT, NPROJ, NPROJP);

  // 2) in-projection: zx[8192][2184] = u @ W_in
  gemm_bf16_nt<<<dim3(NPROJP / 16, NTOK / 512), 256, 0, stream>>>(u16b, wt, zx, NTOK, NPROJ, NFEAT, NPROJ);

  // 3) dt softplus + causal depthwise convs
  dt_softplus_kernel<<<(NTOK * NHEADS + 255) / 256, 256, 0, stream>>>(zx, dt_bias, dtb);
  dwconv_silu_kernel<<<(NTOK * CONVD + 255) / 256, 256, 0, stream>>>(zx, xconv_w, xconv_b, xcb, HALFD, CONVD);
  dwconv_silu_kernel<<<(NTOK * HALFD + 255) / 256, 256, 0, stream>>>(zx, zconv_w, zconv_b, zcb, 0, HALFD);

  // 4) SSD core
  ssd_chunk_kernel<<<NB * NCHUNK * NHEADS, 256, 0, stream>>>(xcb, dtb, Alog, Dp, csb, stb, yfb);
  ssd_scan_kernel<<<(NB * NHEADS * HEADDIM * NSTATES + 255) / 256, 256, 0, stream>>>(stb, csb, pvb);
  ssd_yoff_kernel<<<NB * NCHUNK * NHEADS, 256, 0, stream>>>(xcb, csb, pvb, yfb);

  // 5) RMS norm + concat -> bf16
  rmsnorm_cat_kernel<<<NTOK, 256, 0, stream>>>(yfb, zcb, norm_w, y16b);

  // 6) out-projection: out[8192][1024] = y @ W_out
  transpose_cvt_kernel<<<(NFEAT * INNERD + 255) / 256, 256, 0, stream>>>(W_out, wot, INNERD, NFEAT, NFEAT);
  gemm_bf16_nt<<<dim3(NFEAT / 16, NTOK / 512), 256, 0, stream>>>(y16b, wot, out, NTOK, NFEAT, INNERD, NFEAT);

  (void)in_sizes; (void)n_in; (void)out_size; (void)ws_size;
}
